// GINConvNet_45981919871649
// MI455X (gfx1250) — matrix operand order repaired and checked by reading, not compiled, not run on hardware
//
#include <hip/hip_runtime.h>
#include <hip/hip_bf16.h>

// ---------------------------------------------------------------------------
// GINConvNet inference for MI455X (gfx1250, wave32, WMMA).
// All dense math runs through v_wmma_f32_16x16x32_f16 (f16 inputs, f32 acc).
// Conv layers are implicit-im2col GEMMs (A gathered on the fly from NLC
// activation buffers or directly from the f16 embedding table). Each wave
// computes a 16x(16*NT) output strip (one A fragment feeds NT WMMAs). The K
// loop processes two K-steps per iteration with independent fragment regs:
// all loads issue first (clauseable, partial waits), then the two WMMA
// groups -- no cross-iteration register rotation (no v_mov churn).
// ---------------------------------------------------------------------------

typedef _Float16 h16;
typedef __attribute__((ext_vector_type(16))) _Float16 v16h;
typedef __attribute__((ext_vector_type(8)))  _Float16 v8h;
typedef __attribute__((ext_vector_type(8)))  float    v8f;

#define DIMC 32
#define NB   512
#define NPGC 64
#define NNODE (NB*NPGC)     // 32768
#define NEDGE (NNODE*8)     // 262144
#define FXDC 78

// ----------------------------- small kernels -------------------------------

// f32 -> f16 with zero padding: dst is [Np x Kp], src is [N x K]
__global__ void k_cvt_pad16(const float* __restrict__ src, h16* __restrict__ dst,
                            int N, int K, int Np, int Kp) {
  int i = blockIdx.x * 256 + threadIdx.x;
  int total = Np * Kp;
  if (i >= total) return;
  int n = i / Kp, k = i % Kp;
  float v = (n < N && k < K) ? src[(long)n * K + k] : 0.f;
  dst[i] = (h16)v;
}

// agg[dst[e], c] += h[src[e], c]   (edge-parallel atomic scatter)
__global__ void k_scatter_add(const float* __restrict__ h, const int* __restrict__ esrc,
                              const int* __restrict__ edst, float* __restrict__ agg,
                              int E, int C) {
  int i = blockIdx.x * 256 + threadIdx.x;
  if (i >= E * C) return;
  int e = i / C, c = i - e * C;
  atomicAdd(&agg[(long)edst[e] * C + c], h[(long)esrc[e] * C + c]);
}

// s16[n,k] = f16( h[n,k] + agg[n,k] ) for k<C, else 0   (dst [N x Cp])
__global__ void k_add_cvt(const float* __restrict__ a, const float* __restrict__ b,
                          h16* __restrict__ out, int N, int C, int Cp) {
  int i = blockIdx.x * 256 + threadIdx.x;
  if (i >= N * Cp) return;
  int n = i / Cp, c = i - n * Cp;
  float v = (c < C) ? (a[(long)n * C + c] + b[(long)n * C + c]) : 0.f;
  out[i] = (h16)v;
}

// g = batchnorm(relu(acc))  (eval-mode BN with running stats, eps=1e-5)
__global__ void k_bn_relu(const float* __restrict__ acc, const float* __restrict__ gam,
                          const float* __restrict__ bet, const float* __restrict__ mu,
                          const float* __restrict__ var, float* __restrict__ out,
                          int N, int C) {
  int i = blockIdx.x * 256 + threadIdx.x;
  if (i >= N * C) return;
  int c = i % C;
  float x = fmaxf(acc[i], 0.f);
  out[i] = (x - mu[c]) * rsqrtf(var[c] + 1e-5f) * gam[c] + bet[c];
}

// pooled[batch[n], c] += g[n, c]
__global__ void k_segsum(const float* __restrict__ g, const int* __restrict__ batch,
                         float* __restrict__ pooled, int N, int C) {
  int i = blockIdx.x * 256 + threadIdx.x;
  if (i >= N * C) return;
  int n = i / C, c = i - n * C;
  atomicAdd(&pooled[(long)batch[n] * C + c], g[i]);
}

// src: NLC f16 [B, L, Cs]; out: f16 [B, Fp], f[b, c*Lp+lp] = max_t src[b, lp*kk+t, c]
__global__ void k_pool_flat(const h16* __restrict__ src, h16* __restrict__ out,
                            int B, int L, int Cs, int C, int kk, int Lp, int F, int Fp) {
  int i = blockIdx.x * 256 + threadIdx.x;
  if (i >= B * Fp) return;
  int b = i / Fp, j = i - b * Fp;
  h16 r = (h16)0.f;
  if (j < F) {
    int c = j / Lp, lp = j - c * Lp;
    float m = -1e30f;
    for (int t = 0; t < kk; ++t)
      m = fmaxf(m, (float)src[((long)b * L + lp * kk + t) * Cs + c]);
    r = (h16)m;
  }
  out[i] = r;
}

// External-attention gating collapses analytically: softmax over a size-1 axis
// is 1, then /sum over a size-1 axis is 1 => attn==1 => A[c] = sum_j ea_mv[c,j].
// drugs = drugs*0.5 + drugs*sigmoid(A); g likewise (identical gate).
__global__ void k_gate(float* __restrict__ drugs, float* __restrict__ g,
                       const float* __restrict__ mv /* [128,4] */) {
  int i = blockIdx.x * 256 + threadIdx.x;
  if (i >= NB * 128) return;
  int c = i & 127;
  float s = mv[c * 4 + 0] + mv[c * 4 + 1] + mv[c * 4 + 2] + mv[c * 4 + 3];
  s = 1.f / (1.f + expf(-s));
  drugs[i] = drugs[i] * 0.5f + drugs[i] * s;
  g[i]     = g[i] * 0.5f + g[i] * s;
}

// xc16[b, 0:128]=drugs, [128:256]=g, [256:384]=xt   (f16 staging for fc1)
__global__ void k_concat(const float* __restrict__ a, const float* __restrict__ b,
                         const float* __restrict__ c, h16* __restrict__ out) {
  int i = blockIdx.x * 256 + threadIdx.x;
  if (i >= NB * 384) return;
  int bb = i / 384, j = i - bb * 384;
  float v = (j < 128) ? a[bb * 128 + j]
          : (j < 256) ? b[bb * 128 + (j - 128)]
                      : c[bb * 128 + (j - 256)];
  out[i] = (h16)v;
}

// ----------------------------- WMMA GEMM -----------------------------------
// C[M,N] = act( A[M,K] * W[N,K]^T + bias )  (i.e. x @ W.T + b, as in linear())
// One 16x(16*NT) strip per wave; K looped 64 at a time (two WMMA K-steps) via
// v_wmma_f32_16x16x32_f16; one A fragment shared by NT B fragments.
// A-operand modes:
//   0 PLAIN : A row-major f16, lda (multiple of 32)
//   1 CONV  : implicit im2col from NLC f16 activations [B, Lin, Cs];
//             row m -> (b=m/Lout, l=m%Lout); col ci*8+k -> act[b, l+k, ci]
//   2 EMB   : like CONV but act[b, pos, :] = table[IDX[b*Cin+pos], :] (Lin wide)
// Fragment layouts per CDNA5 ISA 7.12.2 (wave32):
//   A 16x32 f16: lane m=lane%16; halves 0..7 = K off..off+7, 8..15 = K off+16..,
//                off = (lane<16 ? 0 : 8)
//   B 32x16 f16: lane n=lane%16; same K layout as A (off..off+7, off+16..off+23)
//   C/D f32:     lane col=lane%16; VGPR r = row (lane<16 ? r : 8+r)
enum { MODE_PLAIN = 0, MODE_CONV = 1, MODE_EMB = 2 };

template <int MODE, int NT>
__global__ __launch_bounds__(128)
void k_wmma_gemm(const h16* __restrict__ A, const int* __restrict__ IDX,
                 const h16* __restrict__ W, const float* __restrict__ bias,
                 float* __restrict__ Cf, h16* __restrict__ Ch,
                 int M, int Kp, int lda,
                 int Lout, int Lin, int Cin, int Cs,
                 int Npad, int Nstore, int ldc, int relu) {
  const int lane = threadIdx.x;                       // 0..31
  const int strip = blockIdx.x * blockDim.y + threadIdx.y;
  const int strips = Npad / (16 * NT);
  const int tm = blockIdx.y;
  if (strip >= strips) return;                        // wave-uniform
  const int tn0 = strip * NT;

  const int m = tm * 16 + (lane & 15);
  const int offA = (lane < 16) ? 0 : 8;
  const int offB = (lane < 16) ? 0 : 16;

  const h16* wrow = W + (long)(tn0 * 16 + (lane & 15)) * Kp + offA;
  const h16* arow = nullptr;
  int b = 0, l = 0;
  if (MODE == MODE_PLAIN) {
    arow = A + (long)m * lda + offA;
  } else {
    b = m / Lout;
    l = m - b * Lout;
  }

  auto loadA = [&](int kb) -> v16h {
    v16h a;
    if (MODE == MODE_PLAIN) {
      v8h a0 = *(const v8h*)(arow + kb);
      v8h a1 = *(const v8h*)(arow + kb + 16);
#pragma unroll
      for (int j = 0; j < 8; ++j) { a[j] = a0[j]; a[8 + j] = a1[j]; }
    } else if (MODE == MODE_CONV) {
      const int kk0 = kb + offA;
      const int ci0 = kk0 >> 3, ci1 = (kk0 + 16) >> 3;   // K col = ci*8 + k
      const h16* bse = A + ((long)b * Lin + l) * Cs;
#pragma unroll
      for (int j = 0; j < 8; ++j) {
        a[j]     = (ci0 < Cin) ? bse[(long)j * Cs + ci0] : (h16)0.f;
        a[8 + j] = (ci1 < Cin) ? bse[(long)j * Cs + ci1] : (h16)0.f;
      }
    } else {                                             // MODE_EMB
      const int kk0 = kb + offA;
      const int ci0 = kk0 >> 3, ci1 = (kk0 + 16) >> 3;
      const int id0 = (ci0 < Cin) ? IDX[(long)b * Cin + ci0] : -1;
      const int id1 = (ci1 < Cin) ? IDX[(long)b * Cin + ci1] : -1;
#pragma unroll
      for (int j = 0; j < 8; ++j) {
        a[j]     = (id0 >= 0) ? A[(long)id0 * Lin + l + j] : (h16)0.f;
        a[8 + j] = (id1 >= 0) ? A[(long)id1 * Lin + l + j] : (h16)0.f;
      }
    }
    return a;
  };
  auto loadW = [&](int kb, int t) -> v16h {
    const h16* wr = wrow + (long)t * 16 * Kp + kb;
    v8h w0 = *(const v8h*)(wr);
    v8h w1 = *(const v8h*)(wr + 16);
    v16h w;
#pragma unroll
    for (int j = 0; j < 8; ++j) { w[j] = w0[j]; w[8 + j] = w1[j]; }
    return w;
  };

  v8f acc[NT] = {};

  int kb = 0;
  // main loop: two K-steps per iteration; all loads issue before the WMMAs so
  // the second group's loads fly behind the first group's matrix ops.
  for (; kb + 64 <= Kp; kb += 64) {
    v16h a0 = loadA(kb);
    v16h w0[NT];
#pragma unroll
    for (int t = 0; t < NT; ++t) w0[t] = loadW(kb, t);
    v16h a1 = loadA(kb + 32);
    v16h w1[NT];
#pragma unroll
    for (int t = 0; t < NT; ++t) w1[t] = loadW(kb + 32, t);
#pragma unroll
    for (int t = 0; t < NT; ++t)
      acc[t] = __builtin_amdgcn_wmma_f32_16x16x32_f16(false, a0, false, w0[t],
                                                      (short)0, acc[t], false, false);
#pragma unroll
    for (int t = 0; t < NT; ++t)
      acc[t] = __builtin_amdgcn_wmma_f32_16x16x32_f16(false, a1, false, w1[t],
                                                      (short)0, acc[t], false, false);
  }
  if (kb < Kp) {   // uniform tail for Kp % 64 == 32
    v16h a0 = loadA(kb);
    v16h w0[NT];
#pragma unroll
    for (int t = 0; t < NT; ++t) w0[t] = loadW(kb, t);
#pragma unroll
    for (int t = 0; t < NT; ++t)
      acc[t] = __builtin_amdgcn_wmma_f32_16x16x32_f16(false, a0, false, w0[t],
                                                      (short)0, acc[t], false, false);
  }

  // epilogue (f32 C/D layout): col=lane%16, rows rowbase..rowbase+7
  const int rowbase = tm * 16 + ((lane < 16) ? 0 : 8);
#pragma unroll
  for (int t = 0; t < NT; ++t) {
    const int col = (tn0 + t) * 16 + (lane & 15);
    if (col < Nstore) {
      const float bv = bias ? bias[col] : 0.f;
#pragma unroll
      for (int r = 0; r < 8; ++r) {
        float v = acc[t][r] + bv;
        if (relu) v = fmaxf(v, 0.f);
        const long o = (long)(rowbase + r) * ldc + col;
        if (Cf) Cf[o] = v;
        if (Ch) Ch[o] = (h16)v;
      }
    }
  }
}

// ----------------------------- host side -----------------------------------

static inline int cdiv(int a, int b) { return (a + b - 1) / b; }

template <int MODE, int NT>
static void gemm_launch(hipStream_t s, const h16* A, const int* idx, const h16* W,
                        const float* bias, float* Cf, h16* Ch, int M, int Kp, int lda,
                        int Lout, int Lin, int Cin, int Cs,
                        int Npad, int Nstore, int ldc, int relu) {
  dim3 blk(32, 4, 1);
  dim3 grd(cdiv(Npad / (16 * NT), 4), M / 16, 1);
  k_wmma_gemm<MODE, NT><<<grd, blk, 0, s>>>(A, idx, W, bias, Cf, Ch, M, Kp, lda,
                                            Lout, Lin, Cin, Cs, Npad, Nstore, ldc, relu);
}

template <int MODE>
static void gemm(hipStream_t s, const h16* A, const int* idx, const h16* W,
                 const float* bias, float* Cf, h16* Ch, int M, int Kp, int lda,
                 int Lout, int Lin, int Cin, int Cs,
                 int Npad, int Nstore, int ldc, int relu) {
  const int tiles = Npad / 16;
  if ((tiles & 3) == 0)
    gemm_launch<MODE, 4>(s, A, idx, W, bias, Cf, Ch, M, Kp, lda, Lout, Lin, Cin, Cs, Npad, Nstore, ldc, relu);
  else if ((tiles & 1) == 0)
    gemm_launch<MODE, 2>(s, A, idx, W, bias, Cf, Ch, M, Kp, lda, Lout, Lin, Cin, Cs, Npad, Nstore, ldc, relu);
  else
    gemm_launch<MODE, 1>(s, A, idx, W, bias, Cf, Ch, M, Kp, lda, Lout, Lin, Cin, Cs, Npad, Nstore, ldc, relu);
}

static void cvt(hipStream_t s, const float* src, h16* dst, int N, int K, int Np, int Kp) {
  int total = Np * Kp;
  k_cvt_pad16<<<cdiv(total, 256), 256, 0, s>>>(src, dst, N, K, Np, Kp);
}

// Input indices: setup_inputs() dict order with the nested params dict
// flattened in insertion order (x, 80 params, edge_index, batch, target, drug).
extern "C" void kernel_launch(void* const* d_in, const int* in_sizes, int n_in,
                              void* d_out, int out_size, void* d_ws, size_t ws_size,
                              hipStream_t stream) {
  (void)in_sizes; (void)n_in; (void)out_size; (void)ws_size;
  const float* x      = (const float*)d_in[0];
  const int*   edge   = (const int*)d_in[81];   // [2,E]: src row then dst row
  const int*   batch  = (const int*)d_in[82];
  const int*   target = (const int*)d_in[83];
  const int*   drug   = (const int*)d_in[84];
  auto F = [&](int i) { return (const float*)d_in[i]; };

  // bump allocator over workspace
  char* base = (char*)d_ws;
  size_t off = 0;
  auto ab = [&](size_t bytes) -> void* {
    off = (off + 255) & ~(size_t)255;
    void* p = base + off; off += bytes; return p;
  };
  auto ah = [&](size_t n) { return (h16*)ab(n * 2); };
  auto af = [&](size_t n) { return (float*)ab(n * 4); };

  // f16 weight staging (padded: rows->mult of 16, cols->mult of 32)
  h16* gw1[5]; h16* gw2[5];
  gw1[0] = ah(32 * 96);
  for (int i = 1; i < 5; ++i) gw1[i] = ah(32 * 32);
  for (int i = 0; i < 5; ++i) gw2[i] = ah(32 * 32);
  h16* w_fc1xd = ah(128 * 32);
  h16* w_cxt1 = ah(32 * 8000);  h16* w_cxt2 = ah(64 * 256);
  h16* w_cxt3 = ah(96 * 512);   h16* w_cxt4 = ah(128 * 768);
  h16* w_f1xt = ah(1024 * 4224); h16* w_f2xt = ah(128 * 1024);
  h16* w_cxds1 = ah(32 * 800);  h16* w_cxds2 = ah(64 * 224);
  h16* w_cxds3 = ah(96 * 416);  h16* w_cxds4 = ah(128 * 608);
  h16* w_f1xds = ah(1024 * 5024); h16* w_f2xds = ah(128 * 1024);
  h16* w_fc1 = ah(1024 * 384); h16* w_fc2 = ah(256 * 1024); h16* w_outw = ah(16 * 256);
  h16* e_xt = ah(26 * 128); h16* e_xds = ah(65 * 128);

  // f16 activation staging (NLC, strides = padded Cout of producer)
  h16* t1 = ah((size_t)NB * 121 * 32);  h16* t2 = ah((size_t)NB * 114 * 64);
  h16* t3 = ah((size_t)NB * 107 * 96);  h16* t4 = ah((size_t)NB * 100 * 128);
  h16* ft = ah((size_t)NB * 4224);      h16* axt1 = ah((size_t)NB * 1024);
  h16* dd1 = ah((size_t)NB * 121 * 32); h16* dd2 = ah((size_t)NB * 114 * 64);
  h16* dd3 = ah((size_t)NB * 107 * 96); h16* dd4 = ah((size_t)NB * 100 * 128);
  h16* fd = ah((size_t)NB * 5024);      h16* axds1 = ah((size_t)NB * 1024);
  h16* s16 = ah((size_t)NNODE * 96);    h16* z16 = ah((size_t)NNODE * 32);
  h16* pooled16 = ah(NB * 32);          h16* xc16 = ah(NB * 384);
  h16* afc1 = ah((size_t)NB * 1024);    h16* afc2 = ah((size_t)NB * 256);

  // f32 scratch
  float* agg = af((size_t)NNODE * FXDC);
  float* gacc = af((size_t)NNODE * 32);
  float* gcur = af((size_t)NNODE * 32);
  float* pooledf = af(NB * 32);
  float* g512 = af(NB * 128); float* xt512 = af(NB * 128); float* drugs512 = af(NB * 128);

  // --- weight conversion (every call; deterministic, no caching) ---
  cvt(stream, F(1 + 0 * 8), gw1[0], 32, FXDC, 32, 96);
  for (int i = 1; i < 5; ++i) cvt(stream, F(1 + i * 8), gw1[i], 32, 32, 32, 32);
  for (int i = 0; i < 5; ++i) cvt(stream, F(1 + i * 8 + 2), gw2[i], 32, 32, 32, 32);
  cvt(stream, F(41), w_fc1xd, 128, 32, 128, 32);
  cvt(stream, F(43), e_xt, 26, 128, 26, 128);
  cvt(stream, F(44), w_cxt1, 32, 8000, 32, 8000);
  cvt(stream, F(46), w_cxt2, 64, 256, 64, 256);
  cvt(stream, F(48), w_cxt3, 96, 512, 96, 512);
  cvt(stream, F(50), w_cxt4, 128, 768, 128, 768);
  cvt(stream, F(52), w_f1xt, 1024, 4224, 1024, 4224);
  cvt(stream, F(54), w_f2xt, 128, 1024, 128, 1024);
  cvt(stream, F(56), e_xds, 65, 128, 65, 128);
  cvt(stream, F(57), w_cxds1, 25, 800, 32, 800);
  cvt(stream, F(59), w_cxds2, 50, 200, 64, 224);
  cvt(stream, F(61), w_cxds3, 75, 400, 96, 416);
  cvt(stream, F(63), w_cxds4, 100, 600, 128, 608);
  cvt(stream, F(65), w_f1xds, 1024, 5000, 1024, 5024);
  cvt(stream, F(67), w_f2xds, 128, 1024, 128, 1024);
  cvt(stream, F(75), w_fc1, 1024, 384, 1024, 384);
  cvt(stream, F(77), w_fc2, 256, 1024, 256, 1024);
  cvt(stream, F(79), w_outw, 1, 256, 16, 256);

  // --- graph branch: 5x (scatter-agg -> MLP(WMMA) -> ReLU -> BN) ---
  const float* hcur = x; int C = FXDC, Cp = 96;
  for (int i = 0; i < 5; ++i) {
    const int pb = 1 + i * 8;
    hipMemsetAsync(agg, 0, (size_t)NNODE * C * 4, stream);
    k_scatter_add<<<cdiv(NEDGE * C, 256), 256, 0, stream>>>(hcur, edge, edge + NEDGE, agg, NEDGE, C);
    k_add_cvt<<<cdiv(NNODE * Cp, 256), 256, 0, stream>>>(hcur, agg, s16, NNODE, C, Cp);
    gemm<MODE_PLAIN>(stream, s16, nullptr, gw1[i], F(pb + 1), nullptr, z16,
                     NNODE, Cp, Cp, 1, 1, 1, 1, 32, 32, 32, 1);
    gemm<MODE_PLAIN>(stream, z16, nullptr, gw2[i], F(pb + 3), gacc, nullptr,
                     NNODE, 32, 32, 1, 1, 1, 1, 32, 32, 32, 0);
    k_bn_relu<<<cdiv(NNODE * 32, 256), 256, 0, stream>>>(gacc, F(pb + 4), F(pb + 5),
                                                         F(pb + 6), F(pb + 7), gcur, NNODE, 32);
    hcur = gcur; C = 32; Cp = 32;
  }
  hipMemsetAsync(pooledf, 0, (size_t)NB * 32 * 4, stream);
  k_segsum<<<cdiv(NNODE * 32, 256), 256, 0, stream>>>(gcur, batch, pooledf, NNODE, 32);
  cvt(stream, pooledf, pooled16, NB, 32, NB, 32);
  gemm<MODE_PLAIN>(stream, pooled16, nullptr, w_fc1xd, F(42), g512, nullptr,
                   NB, 32, 32, 1, 1, 1, 1, 128, 128, 128, 1);

  // --- protein target branch: 4 convs (implicit im2col WMMA) + pool + MLP ---
  gemm<MODE_EMB>(stream, e_xt, target, w_cxt1, F(45), nullptr, t1,
                 NB * 121, 8000, 0, 121, 128, 1000, 0, 32, 32, 32, 1);
  gemm<MODE_CONV>(stream, t1, nullptr, w_cxt2, F(47), nullptr, t2,
                  NB * 114, 256, 0, 114, 121, 32, 32, 64, 64, 64, 1);
  gemm<MODE_CONV>(stream, t2, nullptr, w_cxt3, F(49), nullptr, t3,
                  NB * 107, 512, 0, 107, 114, 64, 64, 96, 96, 96, 1);
  gemm<MODE_CONV>(stream, t3, nullptr, w_cxt4, F(51), nullptr, t4,
                  NB * 100, 768, 0, 100, 107, 96, 96, 128, 128, 128, 1);
  k_pool_flat<<<cdiv(NB * 4224, 256), 256, 0, stream>>>(t4, ft, NB, 100, 128, 128, 3, 33, 4224, 4224);
  gemm<MODE_PLAIN>(stream, ft, nullptr, w_f1xt, F(53), nullptr, axt1,
                   NB, 4224, 4224, 1, 1, 1, 1, 1024, 1024, 1024, 1);
  gemm<MODE_PLAIN>(stream, axt1, nullptr, w_f2xt, F(55), xt512, nullptr,
                   NB, 1024, 1024, 1, 1, 1, 1, 128, 128, 128, 0);

  // --- drug SMILES branch ---
  gemm<MODE_EMB>(stream, e_xds, drug, w_cxds1, F(58), nullptr, dd1,
                 NB * 121, 800, 0, 121, 128, 100, 0, 32, 25, 32, 1);
  gemm<MODE_CONV>(stream, dd1, nullptr, w_cxds2, F(60), nullptr, dd2,
                  NB * 114, 224, 0, 114, 121, 25, 32, 64, 50, 64, 1);
  gemm<MODE_CONV>(stream, dd2, nullptr, w_cxds3, F(62), nullptr, dd3,
                  NB * 107, 416, 0, 107, 114, 50, 64, 96, 75, 96, 1);
  gemm<MODE_CONV>(stream, dd3, nullptr, w_cxds4, F(64), nullptr, dd4,
                  NB * 100, 608, 0, 100, 107, 75, 96, 128, 100, 128, 1);
  k_pool_flat<<<cdiv(NB * 5024, 256), 256, 0, stream>>>(dd4, fd, NB, 100, 128, 100, 2, 50, 5000, 5024);
  gemm<MODE_PLAIN>(stream, fd, nullptr, w_f1xds, F(66), nullptr, axds1,
                   NB, 5024, 5024, 1, 1, 1, 1, 1024, 1024, 1024, 1);
  gemm<MODE_PLAIN>(stream, axds1, nullptr, w_f2xds, F(68), drugs512, nullptr,
                   NB, 1024, 1024, 1, 1, 1, 1, 128, 128, 128, 0);

  // --- attention gate (analytically collapsed), concat, head MLP ---
  k_gate<<<cdiv(NB * 128, 256), 256, 0, stream>>>(drugs512, g512, F(70));
  k_concat<<<cdiv(NB * 384, 256), 256, 0, stream>>>(drugs512, g512, xt512, xc16);
  gemm<MODE_PLAIN>(stream, xc16, nullptr, w_fc1, F(76), nullptr, afc1,
                   NB, 384, 384, 1, 1, 1, 1, 1024, 1024, 1024, 1);
  gemm<MODE_PLAIN>(stream, afc1, nullptr, w_fc2, F(78), nullptr, afc2,
                   NB, 1024, 1024, 1, 1, 1, 1, 256, 256, 256, 1);
  gemm<MODE_PLAIN>(stream, afc2, nullptr, w_outw, F(80), (float*)d_out, nullptr,
                   NB, 256, 256, 1, 1, 1, 1, 16, 1, 1, 0);
}
